// PLabel_75591424409656
// MI455X (gfx1250) — matrix operand order repaired and checked by reading, not compiled, not run on hardware
//
#include <hip/hip_runtime.h>

typedef __attribute__((ext_vector_type(16))) __bf16 bfrag;
typedef __attribute__((ext_vector_type(8)))  float  v8f;

#define BSZ   256
#define TT    64
#define CC    18
#define DIN   64
#define DD    128
#define NG    384           // 3*D
#define NTILE 24            // NG/16
#define NBLK  256
#define NTHR  128

#define WMMA(a,b,c) __builtin_amdgcn_wmma_f32_16x16x32_bf16(false,(a),false,(b),(short)0,(c),false,false)

union FragU { bfrag v; uint4 q[2]; };

__device__ __forceinline__ unsigned short f2bf(float f){
  union { float f; unsigned u; } v; v.f = f;
  unsigned r = v.u + 0x7fffu + ((v.u >> 16) & 1u);
  return (unsigned short)(r >> 16);
}
__device__ __forceinline__ float bf2f(unsigned short h){
  union { unsigned u; float f; } v; v.u = ((unsigned)h) << 16;
  return v.f;
}
__device__ __forceinline__ float sigm(float x){ return 1.f / (1.f + expf(-x)); }

// ---------------- prep kernels ----------------

// Combined input weight: Mw[d][k] = sum_e wih[k][e] * fc_w[e][d]  (K=64 x N=384 B-matrix)
// written directly in WMMA B-fragment lane layout.
__global__ void prep_gi(const float* __restrict__ wih, const float* __restrict__ fc_w,
                        unsigned short* __restrict__ giFrag){
  int id = blockIdx.x * blockDim.x + threadIdx.x;
  if (id >= DIN * NG) return;
  int d = id & 63, k = id >> 6;
  float m = 0.f;
  for (int e = 0; e < DD; ++e) m += wih[k*DD + e] * fc_w[e*DIN + d];
  int kt = d >> 5, dd = d & 31;
  int lane = (((dd >> 3) & 1) << 4) | (k & 15);
  int j = (dd & 7) | ((dd >> 4) << 3);
  int nt = k >> 4;
  giFrag[((kt*NTILE + nt)*32 + lane)*16 + j] = f2bf(m);
}

// whh^T as B-matrix (K=128 x N=384), pre-swizzled fragment layout.
__global__ void prep_whh(const float* __restrict__ whh, unsigned short* __restrict__ whhFrag){
  int id = blockIdx.x * blockDim.x + threadIdx.x;
  if (id >= DD * NG) return;
  int d = id & 127, k = id >> 7;
  float val = whh[k*DD + d];
  int kt = d >> 5, dd = d & 31;
  int lane = (((dd >> 3) & 1) << 4) | (k & 15);
  int j = (dd & 7) | ((dd >> 4) << 3);
  int nt = k >> 4;
  whhFrag[((kt*NTILE + nt)*32 + lane)*16 + j] = f2bf(val);
}

__global__ void prep_misc(const float* __restrict__ wih, const float* __restrict__ fc_b,
                          const float* __restrict__ bih, const float* __restrict__ bhh,
                          const float* __restrict__ w, const float* __restrict__ emb,
                          const float* __restrict__ pred_w,
                          float* __restrict__ biasRZ, float* __restrict__ biasNi,
                          float* __restrict__ biasNh, float* __restrict__ swg,
                          float* __restrict__ ep){
  int id = threadIdx.x;
  if (id < 256) {
    float s = bih[id] + bhh[id];
    for (int e = 0; e < DD; ++e) s += fc_b[e] * wih[id*DD + e];
    biasRZ[id] = s;
  } else if (id < 384) {
    int k = id, kk = id - 256;
    float s = bih[k];
    for (int e = 0; e < DD; ++e) s += fc_b[e] * wih[k*DD + e];
    biasNi[kk] = s;
    biasNh[kk] = bhh[k];
  } else if (id < 384 + CC) {
    swg[id - 384] = 1.f / (1.f + expf(-w[id - 384]));
  } else if (id >= 402 && id < 404) {
    int r = id - 402;
    float s = 0.f;
    for (int d = 0; d < DD; ++d) s += emb[r*DD + d] * pred_w[d];
    ep[r] = s;
  }
}

// ---------------- main persistent GRU kernel ----------------

__device__ __forceinline__ bfrag loadA(const unsigned short* base, int stride, int mt, int kt, int lane){
  int r  = mt*16 + (lane & 15);
  int kb = kt*32 + ((lane >> 4) << 3);
  FragU u;
  u.q[0] = *(const uint4*)(base + r*stride + kb);
  u.q[1] = *(const uint4*)(base + r*stride + kb + 16);
  return u.v;
}

__device__ __forceinline__ void gridbar(int* cnt, int tid){
  __syncthreads();
  if (tid == 0) {
    __threadfence();
    atomicAdd(cnt, 1);
    volatile int* v = cnt;
    while (*v < NBLK) { __builtin_amdgcn_s_sleep(2); }
  }
  __syncthreads();
  __threadfence();
}

__global__ __launch_bounds__(NTHR, 1) void gru_main(
    const float* __restrict__ x,
    const unsigned short* __restrict__ whhFrag,
    const unsigned short* __restrict__ giFrag,
    const float* __restrict__ biasRZ, const float* __restrict__ biasNiG,
    const float* __restrict__ biasNhG,
    const float* __restrict__ swgG, const float* __restrict__ epG,
    const float* __restrict__ il_w1, const float* __restrict__ il_b1,
    const float* __restrict__ il_w2, const float* __restrict__ il_b2,
    const float* __restrict__ pred_w, const float* __restrict__ pred_b,
    float* __restrict__ llA, float* __restrict__ llB,
    int* __restrict__ counters,
    float* __restrict__ out)
{
  __shared__ __align__(16) unsigned short lds_whh[4*NTILE*32*16]; // 96 KB
  __shared__ __align__(16) unsigned short lds_h[32*DD];           // 8 KB
  __shared__ __align__(16) unsigned short lds_x[32*DIN];          // 4 KB
  __shared__ float s_predw[DD];
  __shared__ float s_swg[CC];
  __shared__ float s_hp[32];
  __shared__ float s_red[NTHR];

  const int tid   = threadIdx.x;
  const int lane  = tid & 31;
  const int wid   = tid >> 5;
  const int b     = blockIdx.x;
  const int wbase = wid * 32;

  // stage whh fragments into LDS (96 KB, reused 64 times)
  {
    const uint4* src = (const uint4*)whhFrag;
    uint4* dst = (uint4*)lds_whh;
    for (int i = tid; i < (4*NTILE*32*16)/8; i += NTHR) dst[i] = src[i];
  }
  if (tid < DD) s_predw[tid] = pred_w[tid];
  if (tid < CC) s_swg[tid]   = swgG[tid];
  for (int i = tid; i < 32*DD; i += NTHR) lds_h[i] = 0;
  for (int i = tid; i < (32-CC)*DIN; i += NTHR) lds_x[CC*DIN + i] = 0;

  // per-wave gate bias registers (column = wbase + ntl*16 + (lane&15))
  float bR[2], bZ[2], bNi[2], bNh[2];
  #pragma unroll
  for (int ntl = 0; ntl < 2; ++ntl) {
    int c = wbase + ntl*16 + (lane & 15);
    bR[ntl]  = biasRZ[c];
    bZ[ntl]  = biasRZ[DD + c];
    bNi[ntl] = biasNiG[c];
    bNh[ntl] = biasNhG[c];
  }
  const float ep0 = epG[0], ep1 = epG[1];
  const float pbv = pred_b[0];
  const float b2v = il_b2[0];

  v8f hs[2][2];
  #pragma unroll
  for (int a2 = 0; a2 < 2; ++a2)
    #pragma unroll
    for (int n2 = 0; n2 < 2; ++n2)
      #pragma unroll
      for (int i = 0; i < 8; ++i) hs[a2][n2][i] = 0.f;

  float outrun = -1e30f;
  float* llIn  = llA;
  float* llOut = llB;
  int barIdx = 0;

  __syncthreads();

  for (int t = 0; t < TT; ++t) {
    // ---- stage x rows 0..17 into LDS as bf16 ----
    for (int i = tid; i < CC*DIN; i += NTHR) {
      int c = i >> 6, d = i & 63;
      float v = x[(((size_t)b*TT + t)*CC + c)*DIN + d];
      lds_x[c*DIN + d] = f2bf(v);
    }
    if (t + 1 < TT) {
      const float* nx = x + (((size_t)b*TT + (t+1))*CC)*DIN;
      __builtin_prefetch(nx + tid*8, 0, 1);
    }
    __syncthreads();

    // ---- A fragments ----
    bfrag ax[2][2], ah[2][4];
    #pragma unroll
    for (int mt = 0; mt < 2; ++mt) {
      #pragma unroll
      for (int kt = 0; kt < 2; ++kt) ax[mt][kt] = loadA(lds_x, DIN, mt, kt, lane);
      #pragma unroll
      for (int kt = 0; kt < 4; ++kt) ah[mt][kt] = loadA(lds_h, DD, mt, kt, lane);
    }

    // ---- accumulators ----
    v8f accR[2][2], accZ[2][2], accGI[2][2], accGH[2][2];
    #pragma unroll
    for (int a2 = 0; a2 < 2; ++a2)
      #pragma unroll
      for (int n2 = 0; n2 < 2; ++n2)
        #pragma unroll
        for (int i = 0; i < 8; ++i) {
          accR[a2][n2][i] = 0.f; accZ[a2][n2][i] = 0.f;
          accGI[a2][n2][i] = 0.f; accGH[a2][n2][i] = 0.f;
        }

    // ---- gi = x @ Mw  (K=64, B from global/L2) ----
    #pragma unroll
    for (int kt = 0; kt < 2; ++kt) {
      #pragma unroll
      for (int g = 0; g < 3; ++g) {
        #pragma unroll
        for (int ntl = 0; ntl < 2; ++ntl) {
          int nt = g*8 + wid*2 + ntl;
          FragU u;
          const uint4* p = (const uint4*)(giFrag + ((size_t)(kt*NTILE + nt)*32 + lane)*16);
          u.q[0] = p[0]; u.q[1] = p[1];
          v8f *t0, *t1;
          if (g == 0)      { t0 = &accR[0][ntl];  t1 = &accR[1][ntl];  }
          else if (g == 1) { t0 = &accZ[0][ntl];  t1 = &accZ[1][ntl];  }
          else             { t0 = &accGI[0][ntl]; t1 = &accGI[1][ntl]; }
          *t0 = WMMA(ax[0][kt], u.v, *t0);
          *t1 = WMMA(ax[1][kt], u.v, *t1);
        }
      }
    }

    // ---- gh = h @ whh^T  (K=128, B from LDS) ----
    #pragma unroll
    for (int kt = 0; kt < 4; ++kt) {
      #pragma unroll
      for (int g = 0; g < 3; ++g) {
        #pragma unroll
        for (int ntl = 0; ntl < 2; ++ntl) {
          int nt = g*8 + wid*2 + ntl;
          FragU u;
          const uint4* p = (const uint4*)(lds_whh + ((kt*NTILE + nt)*32 + lane)*16);
          u.q[0] = p[0]; u.q[1] = p[1];
          v8f *t0, *t1;
          if (g == 0)      { t0 = &accR[0][ntl];  t1 = &accR[1][ntl];  }
          else if (g == 1) { t0 = &accZ[0][ntl];  t1 = &accZ[1][ntl];  }
          else             { t0 = &accGH[0][ntl]; t1 = &accGH[1][ntl]; }
          *t0 = WMMA(ah[0][kt], u.v, *t0);
          *t1 = WMMA(ah[1][kt], u.v, *t1);
        }
      }
    }

    __syncthreads();   // all reads of lds_h complete

    // ---- gate math, h update (wave owns its 32 hidden columns) ----
    #pragma unroll
    for (int mt = 0; mt < 2; ++mt) {
      #pragma unroll
      for (int ntl = 0; ntl < 2; ++ntl) {
        #pragma unroll
        for (int i = 0; i < 8; ++i) {
          float r  = sigm(accR[mt][ntl][i] + bR[ntl]);
          float z  = sigm(accZ[mt][ntl][i] + bZ[ntl]);
          float n  = tanhf(accGI[mt][ntl][i] + bNi[ntl] + r*(accGH[mt][ntl][i] + bNh[ntl]));
          float hn = (1.f - z)*n + z*hs[mt][ntl][i];
          hs[mt][ntl][i] = hn;
          int row = mt*16 + ((lane >> 4) << 3) + i;
          lds_h[row*DD + wbase + ntl*16 + (lane & 15)] = f2bf(hn);
        }
      }
    }
    __syncthreads();   // h' ready

    // ---- t==0: initial-label head ----
    if (t == 0) {
      float acc = il_b1[tid];
      for (int c = 0; c < CC; ++c)
        for (int d = 0; d < DD; ++d)
          acc += bf2f(lds_h[c*DD + d]) * il_w1[(size_t)tid*(CC*DD) + c*DD + d];
      float g = 0.5f * acc * (1.f + erff(acc * 0.70710678118f));
      s_red[tid] = g * il_w2[tid];
      __syncthreads();
      if (tid == 0) {
        float s = b2v;
        for (int j = 0; j < NTHR; ++j) s += s_red[j];
        llIn[b] = sigm(s);   // ll0 exchanged globally (torch-tile quirk crosses samples)
      }
      gridbar(&counters[barIdx++], tid);
    }

    // ---- predict head ----
    if (tid < CC) {
      float s = 0.f;
      for (int d = 0; d < DD; ++d) s += bf2f(lds_h[tid*DD + d]) * s_predw[d];
      s_hp[tid] = s;
    }
    __syncthreads();
    if (tid == 0) {
      float m = -1e30f;
      for (int c = 0; c < CC; ++c) {
        int midx = (b*CC + c) & 255;            // faithful torch tile quirk
        float ll  = llIn[midx];
        float mix = (1.f - ll)*ep0 + ll*ep1;
        float lg  = (1.f - s_swg[c])*s_hp[c] + s_swg[c]*mix;
        m = fmaxf(m, lg);
      }
      float cur = sigm(m + pbv);
      outrun = fmaxf(outrun, cur);
      llOut[b] = cur;
    }
    gridbar(&counters[barIdx++], tid);
    { float* tmp = llIn; llIn = llOut; llOut = tmp; }
  }

  if (tid == 0) out[b] = outrun;
}

// ---------------- launch ----------------

extern "C" void kernel_launch(void* const* d_in, const int* in_sizes, int n_in,
                              void* d_out, int out_size, void* d_ws, size_t ws_size,
                              hipStream_t stream) {
  const float* x      = (const float*)d_in[0];
  const float* fc_w   = (const float*)d_in[3];
  const float* fc_b   = (const float*)d_in[4];
  const float* wih    = (const float*)d_in[5];
  const float* whh    = (const float*)d_in[6];
  const float* bih    = (const float*)d_in[7];
  const float* bhh    = (const float*)d_in[8];
  const float* il_w1  = (const float*)d_in[9];
  const float* il_b1  = (const float*)d_in[10];
  const float* il_w2  = (const float*)d_in[11];
  const float* il_b2  = (const float*)d_in[12];
  const float* pred_w = (const float*)d_in[13];
  const float* pred_b = (const float*)d_in[14];
  const float* w      = (const float*)d_in[15];
  const float* emb    = (const float*)d_in[16];

  char* ws = (char*)d_ws;
  unsigned short* whhFrag = (unsigned short*)(ws + 0);        // 98304 B
  unsigned short* giFrag  = (unsigned short*)(ws + 98304);    // 49152 B
  float* biasRZ = (float*)(ws + 147456);                      // 1024 B
  float* biasNi = (float*)(ws + 148480);                      // 512 B
  float* biasNh = (float*)(ws + 148992);                      // 512 B
  float* swg    = (float*)(ws + 149504);                      // 128 B
  float* ep     = (float*)(ws + 149632);                      // 64 B
  float* llA    = (float*)(ws + 149696);                      // 1024 B
  float* llB    = (float*)(ws + 150720);                      // 1024 B
  int*   cnts   = (int*)  (ws + 151744);                      // 130*4 B

  hipMemsetAsync(cnts, 0, (2*TT + 2) * sizeof(int), stream);

  prep_gi <<<(DIN*NG + 127)/128, 128, 0, stream>>>(wih, fc_w, giFrag);
  prep_whh<<<(DD*NG + 127)/128, 128, 0, stream>>>(whh, whhFrag);
  prep_misc<<<1, 512, 0, stream>>>(wih, fc_b, bih, bhh, w, emb, pred_w,
                                   biasRZ, biasNi, biasNh, swg, ep);

  gru_main<<<NBLK, NTHR, 0, stream>>>(x, whhFrag, giFrag,
                                      biasRZ, biasNi, biasNh, swg, ep,
                                      il_w1, il_b1, il_w2, il_b2,
                                      pred_w, pred_b,
                                      llA, llB, cnts,
                                      (float*)d_out);
}
